// LorenzESN_38448547234024
// MI455X (gfx1250) — compile-verified, matrix operand order checked by reading, not run on hardware
//
#include <hip/hip_runtime.h>

// ---------------------------------------------------------------------------
// LorenzESN on gfx1250 (MI455X).
// Sequential 1024-step recurrence; per-step GEMM (32x1568)x(1568x1536) via
// v_wmma_f32_16x16x32_bf16. Weights packed to bf16 once (L2-resident).
// Per wave: 32x16 output (both m-tiles) so each Wcat fragment is loaded once;
// acat staged in LDS per block (100KB <= 320KB WGP LDS; > 64KB WGP$).
// ---------------------------------------------------------------------------

#define SEQ   1024
#define BATCH 32
#define D_IN  32
#define D_OUT 32
#define RSV   1536            // reservoir size R
#define NSYS  (RSV / 3)       // 512 lorenz systems per batch row
#define KTOT  (D_IN + RSV)    // 1568, packed GEMM K

#define SIGMA_C 10.0f
#define RHO_C   28.0f
#define BETA_C  (8.0f / 3.0f)
#define DT_C    0.001f
#define COUP_C  0.1f
#define LEAK_C  1.0f

typedef __attribute__((ext_vector_type(16))) __bf16 v16bf;
typedef __attribute__((ext_vector_type(8)))  float  v8f;

// ---------------- one-time prep ----------------

// Wcat[j, k] = (k < 32) ? W_in[j,k] : COUPLING * W_res[j, k-32]   (bf16, row-major)
__global__ void prep_wcat_kernel(const float* __restrict__ W_in,
                                 const float* __restrict__ W_res,
                                 __bf16* __restrict__ Wcat) {
    int idx = blockIdx.x * blockDim.x + threadIdx.x;   // over RSV*KTOT
    if (idx >= RSV * KTOT) return;
    int j = idx / KTOT, k = idx % KTOT;
    float v = (k < D_IN) ? W_in[j * D_IN + k] : COUP_C * W_res[j * RSV + (k - D_IN)];
    Wcat[idx] = (__bf16)v;
}

// Wout bf16 copy of W_out_w (D_OUT x RSV, row-major)
__global__ void prep_wout_kernel(const float* __restrict__ W_out_w,
                                 __bf16* __restrict__ Wout) {
    int idx = blockIdx.x * blockDim.x + threadIdx.x;   // over D_OUT*RSV
    if (idx >= D_OUT * RSV) return;
    Wout[idx] = (__bf16)W_out_w[idx];
}

__global__ void zero_state_kernel(float* __restrict__ state) {
    int idx = blockIdx.x * blockDim.x + threadIdx.x;   // over BATCH*RSV
    if (idx >= BATCH * RSV) return;
    state[idx] = 0.0f;
}

// ---------------- helpers ----------------

__device__ __forceinline__ void lorenz_deriv(float x, float y, float z,
                                             float& dx, float& dy, float& dz) {
    dx = SIGMA_C * (y - x);
    dy = x * (RHO_C - z) - y;
    dz = x * y - BETA_C * z;
}

// Load a 16x32 bf16 WMMA A/B fragment from a row-major matrix (global or LDS).
// lane L: row = base + L%16, K-halves at kh and kh+16 (each 8 bf16 = 16B).
template <typename PTR>
__device__ __forceinline__ v16bf load_frag_b16(PTR base_ptr, int row_stride,
                                               int row, int kbase, int lane) {
    int r = row + (lane & 15);
    int kh = kbase + ((lane >> 4) << 3);
    auto p = base_ptr + r * row_stride + kh;
    v16bf f;
    uint4* fp = reinterpret_cast<uint4*>(&f);
    fp[0] = *(const uint4*)(p);        // K = kh .. kh+7
    fp[1] = *(const uint4*)(p + 16);   // K = kh+16 .. kh+23
    return f;
}

// y = new @ Wout^T + b for one step.  M=32, N=32, K=1536 -> 4 tiles, 4 waves.
__device__ __forceinline__ void out_gemm_body(const __bf16* __restrict__ newbf,
                                              const __bf16* __restrict__ Wout,
                                              const float* __restrict__ bias,
                                              float* __restrict__ y, int tid) {
    int wave = tid >> 5;               // 0..3
    int lane = tid & 31;
    int m0 = (wave & 1) * 16;
    int n0 = (wave >> 1) * 16;

    v8f c = {};
    for (int kk = 0; kk < RSV / 32; ++kk) {
        v16bf a = load_frag_b16(newbf, RSV, m0, kk * 32, lane);
        v16bf b = load_frag_b16(Wout, RSV, n0, kk * 32, lane);
        c = __builtin_amdgcn_wmma_f32_16x16x32_bf16(false, a, false, b,
                                                    (short)0, c, false, false);
    }
    int n = n0 + (lane & 15);
    int mb = m0 + ((lane >> 4) << 3);
#pragma unroll
    for (int r = 0; r < 8; ++r)
        y[(mb + r) * D_OUT + n] = c[r] + bias[n];
}

// ---------------- per-step kernels ----------------

#define RK4_BLOCKS 68   // 68*256 = 17408 = BATCH*NSYS + BATCH*D_IN

// Fused: blocks 0..67 -> RK4 + pack acat=[u_t|s]; block 68 (4 waves) -> readout
// GEMM for the *previous* step (both depend only on the prior step_gemm).
__global__ void rk4_pack_out_kernel(const float* __restrict__ state,
                                    const float* __restrict__ u_t,       // (BATCH, D_IN)
                                    float* __restrict__ s_f32,           // (BATCH, RSV)
                                    __bf16* __restrict__ acat,           // (BATCH, KTOT)
                                    const __bf16* __restrict__ newbf,    // (BATCH, RSV)
                                    const __bf16* __restrict__ Wout,     // (D_OUT, RSV)
                                    const float* __restrict__ bias,
                                    float* __restrict__ y_prev,          // (BATCH, D_OUT)
                                    int do_out) {
    if (blockIdx.x < RK4_BLOCKS) {
        int tid = blockIdx.x * blockDim.x + threadIdx.x;
        if (tid < BATCH * NSYS) {
            int b = tid / NSYS, i = tid % NSYS;
            const float* sp = state + b * RSV + 3 * i;
            float x = sp[0], y = sp[1], z = sp[2];
            float k1x, k1y, k1z, k2x, k2y, k2z, k3x, k3y, k3z, k4x, k4y, k4z;
            lorenz_deriv(x, y, z, k1x, k1y, k1z);
            k1x *= DT_C; k1y *= DT_C; k1z *= DT_C;
            lorenz_deriv(x + 0.5f * k1x, y + 0.5f * k1y, z + 0.5f * k1z, k2x, k2y, k2z);
            k2x *= DT_C; k2y *= DT_C; k2z *= DT_C;
            lorenz_deriv(x + 0.5f * k2x, y + 0.5f * k2y, z + 0.5f * k2z, k3x, k3y, k3z);
            k3x *= DT_C; k3y *= DT_C; k3z *= DT_C;
            lorenz_deriv(x + k3x, y + k3y, z + k3z, k4x, k4y, k4z);
            k4x *= DT_C; k4y *= DT_C; k4z *= DT_C;
            float nx = x + (k1x + 2.0f * k2x + 2.0f * k3x + k4x) * (1.0f / 6.0f);
            float ny = y + (k1y + 2.0f * k2y + 2.0f * k3y + k4y) * (1.0f / 6.0f);
            float nz = z + (k1z + 2.0f * k2z + 2.0f * k3z + k4z) * (1.0f / 6.0f);
            float* op = s_f32 + b * RSV + 3 * i;
            op[0] = nx; op[1] = ny; op[2] = nz;
            __bf16* ap = acat + b * KTOT + D_IN + 3 * i;
            ap[0] = (__bf16)nx; ap[1] = (__bf16)ny; ap[2] = (__bf16)nz;
        } else {
            int t2 = tid - BATCH * NSYS;            // 0 .. BATCH*D_IN-1
            int b = t2 / D_IN, k = t2 % D_IN;
            acat[b * KTOT + k] = (__bf16)u_t[t2];
        }
    } else {
        if (do_out && threadIdx.x < 128)
            out_gemm_body(newbf, Wout, bias, y_prev, threadIdx.x);
    }
}

// Standalone readout for the final step.
__global__ void out_gemm_kernel(const __bf16* __restrict__ newbf,
                                const __bf16* __restrict__ Wout,
                                const float* __restrict__ bias,
                                float* __restrict__ y) {
    out_gemm_body(newbf, Wout, bias, y, threadIdx.x);
}

// pre = acat @ Wcat^T ; new = (1-LEAK)*s + LEAK*tanh(pre).
// 96 waves (24 blocks x 4 waves); each wave computes a 32x16 tile (both
// m-tiles) so each Wcat fragment is loaded exactly once per step. acat is
// staged into LDS once per block (100,352 B).
__global__ __launch_bounds__(128)
void step_gemm_kernel(const __bf16* __restrict__ acat,   // (BATCH, KTOT)
                      const __bf16* __restrict__ Wcat,   // (RSV, KTOT)
                      const float* __restrict__ s_f32,   // (BATCH, RSV)
                      float* __restrict__ state,         // (BATCH, RSV)
                      __bf16* __restrict__ newbf) {      // (BATCH, RSV)
    __shared__ __bf16 sA[BATCH * KTOT];                  // 100,352 B of LDS

    // cooperative stage: 6272 b128 transfers across 128 threads
    {
        const uint4* src = (const uint4*)acat;
        uint4* dst = (uint4*)sA;
        for (int i = threadIdx.x; i < (BATCH * KTOT * 2) / 16; i += 128)
            dst[i] = src[i];
    }
    __syncthreads();

    int wave = threadIdx.x >> 5;
    int lane = threadIdx.x & 31;
    int n0 = (blockIdx.x * 4 + wave) * 16;   // reservoir-column tile

    v8f c0 = {}, c1 = {};
    for (int kk = 0; kk < KTOT / 32; ++kk) {
        v16bf a0 = load_frag_b16((const __bf16*)sA, KTOT, 0, kk * 32, lane);
        v16bf a1 = load_frag_b16((const __bf16*)sA, KTOT, 16, kk * 32, lane);
        v16bf b  = load_frag_b16(Wcat, KTOT, n0, kk * 32, lane);
        c0 = __builtin_amdgcn_wmma_f32_16x16x32_bf16(false, a0, false, b,
                                                     (short)0, c0, false, false);
        c1 = __builtin_amdgcn_wmma_f32_16x16x32_bf16(false, a1, false, b,
                                                     (short)0, c1, false, false);
    }

    int n = n0 + (lane & 15);
    int mb = (lane >> 4) << 3;
#pragma unroll
    for (int r = 0; r < 8; ++r) {
        int m = mb + r;
        float nv = (1.0f - LEAK_C) * s_f32[m * RSV + n] + LEAK_C * tanhf(c0[r]);
        state[m * RSV + n] = nv;
        newbf[m * RSV + n] = (__bf16)nv;
        int m2 = m + 16;
        float nv2 = (1.0f - LEAK_C) * s_f32[m2 * RSV + n] + LEAK_C * tanhf(c1[r]);
        state[m2 * RSV + n] = nv2;
        newbf[m2 * RSV + n] = (__bf16)nv2;
    }
}

// ---------------- host launch ----------------

extern "C" void kernel_launch(void* const* d_in, const int* in_sizes, int n_in,
                              void* d_out, int out_size, void* d_ws, size_t ws_size,
                              hipStream_t stream) {
    const float* u       = (const float*)d_in[0];   // (SEQ, BATCH, D_IN)
    const float* W_in    = (const float*)d_in[1];   // (RSV, D_IN)
    const float* W_res   = (const float*)d_in[2];   // (RSV, RSV)
    const float* W_out_w = (const float*)d_in[3];   // (D_OUT, RSV)
    const float* W_out_b = (const float*)d_in[4];   // (D_OUT,)
    float* out = (float*)d_out;                     // (SEQ, BATCH, D_OUT)

    // workspace carve-up (all offsets 256B-aligned; total ~5.26 MB)
    char* ws = (char*)d_ws;
    __bf16* Wcat  = (__bf16*)(ws + 0);                       // RSV*KTOT   bf16 = 4,816,896 B
    __bf16* Wout  = (__bf16*)(ws + 4816896);                 // D_OUT*RSV  bf16 =    98,304 B
    float*  state = (float*) (ws + 4915200);                 // BATCH*RSV  f32  =   196,608 B
    float*  s_f32 = (float*) (ws + 5111808);                 // BATCH*RSV  f32  =   196,608 B
    __bf16* acat  = (__bf16*)(ws + 5308416);                 // BATCH*KTOT bf16 =   100,352 B
    __bf16* newbf = (__bf16*)(ws + 5408768);                 // BATCH*RSV  bf16 =    98,304 B

    // one-time packing (re-done every call; deterministic)
    prep_wcat_kernel<<<(RSV * KTOT + 255) / 256, 256, 0, stream>>>(W_in, W_res, Wcat);
    prep_wout_kernel<<<(D_OUT * RSV + 255) / 256, 256, 0, stream>>>(W_out_w, Wout);
    zero_state_kernel<<<(BATCH * RSV + 255) / 256, 256, 0, stream>>>(state);

    for (int t = 0; t < SEQ; ++t) {
        // RK4+pack for step t, fused with readout GEMM for step t-1
        rk4_pack_out_kernel<<<RK4_BLOCKS + 1, 256, 0, stream>>>(
            state, u + (size_t)t * BATCH * D_IN, s_f32, acat,
            newbf, Wout, W_out_b,
            out + (size_t)(t > 0 ? t - 1 : 0) * BATCH * D_OUT, t > 0 ? 1 : 0);
        // 96 n-tiles, 1 wave each computing 32x16 -> 24 blocks x 128 threads
        step_gemm_kernel<<<24, 128, 0, stream>>>(acat, Wcat, s_f32, state, newbf);
    }
    // trailing readout for the final step
    out_gemm_kernel<<<1, 128, 0, stream>>>(newbf, Wout, W_out_b,
                                           out + (size_t)(SEQ - 1) * BATCH * D_OUT);
}
